// RNNModel_2448131358905
// MI455X (gfx1250) — matrix operand order repaired and checked by reading, not compiled, not run on hardware
//
#include <hip/hip_runtime.h>

typedef __attribute__((ext_vector_type(16))) _Float16 v16h;
typedef __attribute__((ext_vector_type(8)))  _Float16 v8h;
typedef __attribute__((ext_vector_type(8)))  float    v8f;
typedef __attribute__((ext_vector_type(4)))  float    v4f;

#define B_TOT  2048
#define T_LEN  512
#define I_DIM  8
#define H_DIM  64
#define LDSROW 72   // halves per LDS row: 64 + 8 pad (144B stride -> conflict-free b128)
#define NTILE  2    // independent batch tiles per workgroup (fills WMMA hazard bubbles)

__device__ __forceinline__ float fast_tanh(float x) {
#if __has_builtin(__builtin_amdgcn_tanhf)
    return __builtin_amdgcn_tanhf(x);          // v_tanh_f32 (gfx1250 transcendental)
#else
    float e = __expf(2.0f * x);
    return 1.0f - 2.0f / (e + 1.0f);
#endif
}

// A-operand (16 rows x K=32 f16) chunk from a row-major f32 weight matrix.
// f16 A layout: lane<16 holds K = kbase+{0..15}; lane>=16 holds K = kbase+{16..31}.
__device__ __forceinline__ v16h load_wA(const float* W, int ld, int row, int kbase, int lane) {
    const int koff = kbase + ((lane & 16) ? 16 : 0);
    const float* p = W + row * ld + koff;
    v16h a;
#pragma unroll
    for (int i = 0; i < 8; ++i) { a[i] = (_Float16)p[i]; a[i + 8] = (_Float16)p[i + 8]; }
    return a;
}

// A-operand for the K=8 input GEMM (W_ih0 is [64][8]); K=8..31 zero-padded.
__device__ __forceinline__ v16h load_wA_k8(const float* W, int row, int lane) {
    v16h a;
#pragma unroll
    for (int i = 0; i < 16; ++i) a[i] = (_Float16)0.0f;
    if (!(lane & 16)) {
        const float* p = W + row * I_DIM;
#pragma unroll
        for (int i = 0; i < 8; ++i) a[i] = (_Float16)p[i];   // K = 0..7
    }
    return a;
}

// B-operand (K=32 x 16 batch) from LDS h buffer laid out [b][j] row-major f16.
// B layout: lane holds column b = lane%16; lanes<16 K = 32c+0..15, lanes>=16 K = 32c+16..31.
__device__ __forceinline__ v16h ldsB(const _Float16* buf, int lane, int kchunk) {
    const _Float16* p = buf + (lane & 15) * LDSROW + 32 * kchunk + ((lane & 16) ? 16 : 0);
    v8h lo = *(const v8h*)(p);        // ds_load_b128
    v8h hi = *(const v8h*)(p + 8);    // ds_load_b128
    return __builtin_shufflevector(lo, hi, 0,1,2,3,4,5,6,7,8,9,10,11,12,13,14,15);
}

// Publish one 16x16 C tile of h (f32, C layout) to LDS [b][j] f16. One ds_store_b128.
// C layout: element r <-> (row = wv*16 + 8*(lane/16) + r, col b = lane%16).
__device__ __forceinline__ void storeH(_Float16* buf, int lane, int wv, const float* h8) {
    v8h hv;
#pragma unroll
    for (int i = 0; i < 8; ++i) hv[i] = (_Float16)h8[i];
    *(v8h*)(buf + (lane & 15) * LDSROW + wv * 16 + ((lane & 16) ? 8 : 0)) = hv;
}

__global__ __launch_bounds__(128) void rnn2_fused(
    const float* __restrict__ x,
    const float* __restrict__ Wih0, const float* __restrict__ Whh0,
    const float* __restrict__ bih0, const float* __restrict__ bhh0,
    const float* __restrict__ Wih1, const float* __restrict__ Whh1,
    const float* __restrict__ bih1, const float* __restrict__ bhh1,
    const float* __restrict__ fcw,  const float* __restrict__ fcb,
    float* __restrict__ out)
{
    __shared__ _Float16 h0buf[NTILE][2][16 * LDSROW];  // per-tile double-buffered h0 (f16)
    __shared__ _Float16 h1buf[NTILE][2][16 * LDSROW];  // per-tile double-buffered h1 (f16)
    __shared__ float    red[NTILE][4][32];             // FC reduction scratch

    const int tid  = threadIdx.x;
    const int lane = tid & 31;
    const int wv   = tid >> 5;                  // wave id 0..3 -> output rows wv*16..wv*16+15
    const int b0   = blockIdx.x * (16 * NTILE); // batch base for this workgroup

    // zero parity-0 state (h_0 = 0) for both tiles
    for (int i = tid; i < 16 * LDSROW; i += 128) {
#pragma unroll
        for (int u = 0; u < NTILE; ++u) {
            h0buf[u][0][i] = (_Float16)0.0f;
            h1buf[u][0][i] = (_Float16)0.0f;
        }
    }

    const int mrow = wv * 16 + (lane & 15);   // this lane's A-operand row (global h index)

    // Preload all weight slices as f16 A operands (resident in VGPRs for all 512 steps)
    v16h a_ih0   = load_wA_k8(Wih0, mrow, lane);
    v16h a_hh0_0 = load_wA(Whh0, H_DIM, mrow, 0,  lane);
    v16h a_hh0_1 = load_wA(Whh0, H_DIM, mrow, 32, lane);
    v16h a_ih1_0 = load_wA(Wih1, H_DIM, mrow, 0,  lane);
    v16h a_ih1_1 = load_wA(Wih1, H_DIM, mrow, 32, lane);
    v16h a_hh1_0 = load_wA(Whh1, H_DIM, mrow, 0,  lane);
    v16h a_hh1_1 = load_wA(Whh1, H_DIM, mrow, 32, lane);

    // Biases broadcast into C layout: element r -> h row hb+r (shared by both tiles)
    const int hb = wv * 16 + ((lane & 16) ? 8 : 0);
    v8f bias0, bias1;
#pragma unroll
    for (int r = 0; r < 8; ++r) {
        bias0[r] = bih0[hb + r] + bhh0[hb + r];
        bias1[r] = bih1[hb + r] + bhh1[hb + r];
    }

    const float* xrow[NTILE];
#pragma unroll
    for (int u = 0; u < NTILE; ++u)
        xrow[u] = x + (size_t)(b0 + u * 16 + (lane & 15)) * T_LEN * I_DIM;
    const bool loader = (lane < 16);

    __syncthreads();

    // Layer-0 recurrent B operands live in registers across iterations: after the
    // barrier of step t they hold h0[t] == the operand for both the layer-1 input
    // GEMM of step t AND the layer-0 recurrence of step t+1.
    v16h h0b0[NTILE], h0b1[NTILE];
#pragma unroll
    for (int u = 0; u < NTILE; ++u) {
        h0b0[u] = ldsB(h0buf[u][0], lane, 0);   // zeros (h_0 = 0)
        h0b1[u] = ldsB(h0buf[u][0], lane, 1);
    }

    float h1v[NTILE][8];
#pragma unroll
    for (int u = 0; u < NTILE; ++u)
#pragma unroll
        for (int r = 0; r < 8; ++r) h1v[u][r] = 0.0f;

#pragma unroll 2
    for (int t = 0; t < T_LEN; ++t) {
        const int p = t & 1;

        // ---- x_t as zero-padded K=32 B operands (only K=0..7 live, lanes<16) ----
        v16h bx[NTILE];
#pragma unroll
        for (int u = 0; u < NTILE; ++u) {
#pragma unroll
            for (int i = 0; i < 16; ++i) bx[u][i] = (_Float16)0.0f;
        }
        if (loader) {
            int tp = (t + 16 < T_LEN) ? (t + 16) : (T_LEN - 1);
#pragma unroll
            for (int u = 0; u < NTILE; ++u) {
                v4f x0 = *(const v4f*)(xrow[u] + t * I_DIM);
                v4f x1 = *(const v4f*)(xrow[u] + t * I_DIM + 4);
#pragma unroll
                for (int i = 0; i < 4; ++i) { bx[u][i] = (_Float16)x0[i]; bx[u][i + 4] = (_Float16)x1[i]; }
                __builtin_prefetch(xrow[u] + tp * I_DIM, 0, 1);   // global_prefetch_b8
            }
        }

        // ---- layer 0 (both tiles interleaved): h0 = tanh(Wih0 x + Whh0 h0_prev + b0) ----
        v8f c0[NTILE];
#pragma unroll
        for (int u = 0; u < NTILE; ++u) {
            c0[u] = bias0;
            c0[u] = __builtin_amdgcn_wmma_f32_16x16x32_f16(false, a_ih0,   false, bx[u],   (short)0, c0[u], false, false);
            c0[u] = __builtin_amdgcn_wmma_f32_16x16x32_f16(false, a_hh0_0, false, h0b0[u], (short)0, c0[u], false, false);
            c0[u] = __builtin_amdgcn_wmma_f32_16x16x32_f16(false, a_hh0_1, false, h0b1[u], (short)0, c0[u], false, false);
        }
#pragma unroll
        for (int u = 0; u < NTILE; ++u) {
            float h0v[8];
#pragma unroll
            for (int r = 0; r < 8; ++r) h0v[r] = fast_tanh(c0[u][r]);
            storeH(h0buf[u][p ^ 1], lane, wv, h0v);
        }
        __syncthreads();   // the ONLY barrier per timestep (drains DScnt first)

        // ---- gather all phase-2 B operands up front so the loads pipeline ----
        v16h h1b0[NTILE], h1b1[NTILE];
#pragma unroll
        for (int u = 0; u < NTILE; ++u) {
            h0b0[u] = ldsB(h0buf[u][p ^ 1], lane, 0);    // full h0[t] (reused next step)
            h0b1[u] = ldsB(h0buf[u][p ^ 1], lane, 1);
            h1b0[u] = ldsB(h1buf[u][p], lane, 0);        // h1[t-1]
            h1b1[u] = ldsB(h1buf[u][p], lane, 1);
        }

        // ---- layer 1 (both tiles interleaved): h1 = tanh(Wih1 h0 + Whh1 h1_prev + b1) ----
        v8f c1[NTILE];
#pragma unroll
        for (int u = 0; u < NTILE; ++u) {
            c1[u] = bias1;
            c1[u] = __builtin_amdgcn_wmma_f32_16x16x32_f16(false, a_ih1_0, false, h0b0[u], (short)0, c1[u], false, false);
            c1[u] = __builtin_amdgcn_wmma_f32_16x16x32_f16(false, a_ih1_1, false, h0b1[u], (short)0, c1[u], false, false);
            c1[u] = __builtin_amdgcn_wmma_f32_16x16x32_f16(false, a_hh1_0, false, h1b0[u], (short)0, c1[u], false, false);
            c1[u] = __builtin_amdgcn_wmma_f32_16x16x32_f16(false, a_hh1_1, false, h1b1[u], (short)0, c1[u], false, false);
        }
#pragma unroll
        for (int u = 0; u < NTILE; ++u) {
#pragma unroll
            for (int r = 0; r < 8; ++r) h1v[u][r] = fast_tanh(c1[u][r]);
            storeH(h1buf[u][p ^ 1], lane, wv, h1v[u]);
        }
        // no second barrier: the next iteration's single barrier separates all
        // cross-wave read/write pairs (double-buffered h0/h1; __syncthreads
        // drains DScnt before s_barrier_signal).
    }

    // ---- FC: out[b] = sum_h h1_last[b,h] * fcw[h] + fcb ----
#pragma unroll
    for (int u = 0; u < NTILE; ++u) {
        float part = 0.0f;
#pragma unroll
        for (int r = 0; r < 8; ++r) part += h1v[u][r] * fcw[hb + r];
        red[u][wv][lane] = part;
    }
    __syncthreads();
    if (wv == 0 && lane < 16) {
#pragma unroll
        for (int u = 0; u < NTILE; ++u) {
            float s = fcb[0];
#pragma unroll
            for (int ww = 0; ww < 4; ++ww) s += red[u][ww][lane] + red[u][ww][lane + 16];
            out[b0 + u * 16 + lane] = s;
        }
    }
}

extern "C" void kernel_launch(void* const* d_in, const int* in_sizes, int n_in,
                              void* d_out, int out_size, void* d_ws, size_t ws_size,
                              hipStream_t stream) {
    (void)in_sizes; (void)n_in; (void)out_size; (void)d_ws; (void)ws_size;
    const float* x    = (const float*)d_in[0];
    const float* Wih0 = (const float*)d_in[1];
    const float* Whh0 = (const float*)d_in[2];
    const float* bih0 = (const float*)d_in[3];
    const float* bhh0 = (const float*)d_in[4];
    const float* Wih1 = (const float*)d_in[5];
    const float* Whh1 = (const float*)d_in[6];
    const float* bih1 = (const float*)d_in[7];
    const float* bhh1 = (const float*)d_in[8];
    const float* fcw  = (const float*)d_in[9];
    const float* fcb  = (const float*)d_in[10];
    float* out = (float*)d_out;

    dim3 grid(B_TOT / (16 * NTILE));   // 64 workgroups, 2 batch tiles each
    dim3 block(128);                   // 4 wave32 (H split across 4 M-tiles)
    rnn2_fused<<<grid, block, 0, stream>>>(x, Wih0, Whh0, bih0, bhh0,
                                           Wih1, Whh1, bih1, bhh1, fcw, fcb, out);
}
